// UnfoldLayer_15556371546957
// MI455X (gfx1250) — matrix operand order repaired and checked by reading, not compiled, not run on hardware
//
#include <hip/hip_runtime.h>
#include <hip/hip_bf16.h>
#include <math.h>

// Problem constants (from reference)
#define IN_C   64
#define H_IMG  256
#define W_IMG  256
#define B_SZ   4
#define N_TOK  1024
#define UP_CH  256
#define M_PTS  3072     // total points
#define DC     256      // DOWN_C

typedef __attribute__((ext_vector_type(16))) _Float16 v16h;
typedef __attribute__((ext_vector_type(8)))  float    v8f;

__device__ __forceinline__ float gelu_exact(float x) {
  return 0.5f * x * (1.0f + erff(x * 0.7071067811865475f));
}

// ---------------------------------------------------------------------------
// Generic WMMA GEMM: Out[b,row,n] = act( A[b,row,:] @ W[:,n] + bias[n] )
//
// Block = 128 threads = 4 waves covering a 64-row x 64-col output tile.
// The 32(K) x 64(N) weight tile is cooperatively staged to LDS *pre-swizzled
// into WMMA B-fragment layout* (fragment-contiguous), so each lane fetches
// its 16-element f16 fragment as one aligned 32-byte LDS read (2x
// ds_load_b128). One A fragment feeds 4 WMMAs per K-step.
// EPI: 0 = bias, 1 = bias+gelu,
// 2 = bias+gelu then softmax-mask-weighted reduction over rects -> Z.
// ---------------------------------------------------------------------------
template<int EPI>
__global__ __launch_bounds__(128) void wmma_gemm_kernel(
    const float* __restrict__ A, long long aStride, int lda,
    const float* __restrict__ Wt, int ldw,
    const float* __restrict__ bias,
    float* __restrict__ Out, long long oStride, int ldo,
    int K, int Ncols,
    const float* __restrict__ msoft, float* __restrict__ Z)
{
  // [subtile j][khalf][col ml][elem e] : fragment-contiguous (32B each)
  __shared__ __align__(32) _Float16 bs[4 * 2 * 16 * 16];
  const int tid  = threadIdx.x;
  const int lane = tid & 31;
  const int wave = tid >> 5;
  const int nb   = blockIdx.x * 64;
  const int r0   = (blockIdx.y * 4 + wave) * 16;
  const int b    = blockIdx.z;
  const int ml   = lane & 15;               // row (A) / col (B) within tile
  const int kh   = lane >> 4;               // K-half select per ISA layout
  const int rem  = (Ncols - nb) >> 4;
  const int nsub = rem < 4 ? rem : 4;       // wave-uniform active 16-col subtiles

  const float* __restrict__ arow =
      A + (size_t)b * (size_t)aStride + (size_t)(r0 + ml) * lda;
  const float2* __restrict__ arow2 = (const float2*)arow;

  v8f acc0 = v8f{0.f,0.f,0.f,0.f,0.f,0.f,0.f,0.f};
  v8f acc1 = acc0, acc2 = acc0, acc3 = acc0;

  // lane's private fragment base within LDS for subtile j: frag(j) + 16 f16
  const int fbase = (kh * 16 + ml) * 16;    // within one subtile's 512 f16

  for (int k0 = 0; k0 < K; k0 += 32) {
    // Cooperative B-tile load: 2048 elements / 128 threads = 16 each.
    // Global side coalesced over columns; LDS side pre-swizzled to
    // B-fragment layout: elem (kk,nn) -> [nn>>4][(kk>>3)&1][nn&15][(kk&7)|((kk>>4)<<3)].
#pragma unroll
    for (int i = 0; i < 16; ++i) {
      int e  = i * 128 + tid;
      int kk = e >> 6, nn = e & 63;
      float wv = (nb + nn < Ncols) ? Wt[(size_t)(k0 + kk) * ldw + nb + nn] : 0.f;
      bs[(((nn >> 4) * 2 + ((kk >> 3) & 1)) * 16 + (nn & 15)) * 16 + ((kk & 7) | ((kk >> 4) << 3))] =
          (_Float16)wv;
    }
    __syncthreads();

    __builtin_prefetch(arow + k0 + 64, 0, 1);   // global_prefetch_b8
    // A fragment (16x32 f16): lanes 0-15 rows M=lane, K {0..7,16..23};
    // lanes 16-31 rows M=lane-16, K {8..15,24..31}; pairs packed per VGPR.
    v16h af;
#pragma unroll
    for (int v = 0; v < 8; ++v) {
      int kb = k0 + kh * 8 + ((v < 4) ? (2 * v) : (16 + 2 * (v - 4)));
      float2 p = arow2[kb >> 1];            // global_load_b64
      af[2 * v]     = (_Float16)p.x;
      af[2 * v + 1] = (_Float16)p.y;
    }

    // 4 WMMAs sharing af; each B fragment is one aligned 32-byte LDS read.
    if (0 < nsub) {
      v16h bf = *(const v16h*)(bs + 0 * 512 + fbase);
      acc0 = __builtin_amdgcn_wmma_f32_16x16x32_f16(false, af, false, bf,
                                                    (short)0, acc0, false, false);
    }
    if (1 < nsub) {
      v16h bf = *(const v16h*)(bs + 1 * 512 + fbase);
      acc1 = __builtin_amdgcn_wmma_f32_16x16x32_f16(false, af, false, bf,
                                                    (short)0, acc1, false, false);
    }
    if (2 < nsub) {
      v16h bf = *(const v16h*)(bs + 2 * 512 + fbase);
      acc2 = __builtin_amdgcn_wmma_f32_16x16x32_f16(false, af, false, bf,
                                                    (short)0, acc2, false, false);
    }
    if (3 < nsub) {
      v16h bf = *(const v16h*)(bs + 3 * 512 + fbase);
      acc3 = __builtin_amdgcn_wmma_f32_16x16x32_f16(false, af, false, bf,
                                                    (short)0, acc3, false, false);
    }
    __syncthreads();
  }

  // -------- epilogue (per active subtile) --------
  auto epilogue = [&](int j, const v8f& acc) {
    if (j >= nsub) return;
    const int   n0  = nb + j * 16;
    const int   n   = n0 + ml;
    const float bia = bias[n];

    if (EPI == 2) {
      // Fused: gelu, weight by softmaxed mask over the 4 rects of each point,
      // reduce 4 consecutive rows -> one z row. 16-row tile == 4 points.
      const int g = n0 >> 6;                // channel group (64 cols each)
      float z0 = 0.f, z1 = 0.f;
#pragma unroll
      for (int r = 0; r < 8; ++r) {
        int   row   = r0 + kh * 8 + r;      // C/D layout: M = r (+8 hi lanes)
        float v     = gelu_exact(acc[r] + bia);
        int   point = row >> 2;
        int   s     = row & 3;
        float w     = msoft[(((size_t)b * M_PTS + point) * 4 + s) * 4 + g];
        if (r < 4) z0 += w * v; else z1 += w * v;
      }
      const int p0 = (r0 >> 2) + kh * 2;
      Z[((size_t)b * M_PTS + p0)     * DC + n] = z0;
      Z[((size_t)b * M_PTS + p0 + 1) * DC + n] = z1;
    } else {
      float* __restrict__ orow = Out + (size_t)b * (size_t)oStride;
#pragma unroll
      for (int r = 0; r < 8; ++r) {
        int   row = r0 + kh * 8 + r;
        float v   = acc[r] + bia;
        if (EPI == 1) v = gelu_exact(v);
        orow[(size_t)row * ldo + n] = v;
      }
    }
  };
  epilogue(0, acc0);
  epilogue(1, acc1);
  epilogue(2, acc2);
  epilogue(3, acc3);
}

// ---------------------------------------------------------------------------
// Point prep: gather y rows from the two up-proj outputs in the concat order,
// compute sigmoid*255 rect coords and softmax (over rects) mask weights.
// ---------------------------------------------------------------------------
__global__ void prep_points_kernel(const float* __restrict__ yi0,
                                   const float* __restrict__ yi1,
                                   float* __restrict__ coords,
                                   float* __restrict__ msoft)
{
  int t = blockIdx.x * blockDim.x + threadIdx.x;
  if (t >= B_SZ * M_PTS) return;
  int b = t / M_PTS;
  int m = t - b * M_PTS;
  // concat order: yi0[0:512], yi1[0:768], yi0[512:768], yi1[768:2304]
  const float* base; int row, rows;
  if (m < 512)       { base = yi0; row = m;       rows = 768;  }
  else if (m < 1280) { base = yi1; row = m - 512; rows = 2304; }
  else if (m < 1536) { base = yi0; row = m - 768; rows = 768;  }
  else               { base = yi1; row = m - 768; rows = 2304; }
  const float* y = base + ((size_t)b * rows + row) * 32;

  float* crd = coords + (size_t)t * 16;
#pragma unroll
  for (int j = 0; j < 16; ++j)
    crd[j] = 255.0f / (1.0f + expf(-y[j]));   // sigmoid * (H-1)

  float* ms = msoft + (size_t)t * 16;
#pragma unroll
  for (int g = 0; g < 4; ++g) {
    float v0 = y[16 + g], v1 = y[20 + g], v2 = y[24 + g], v3 = y[28 + g];
    float mx = fmaxf(fmaxf(v0, v1), fmaxf(v2, v3));
    float e0 = expf(v0 - mx), e1 = expf(v1 - mx);
    float e2 = expf(v2 - mx), e3 = expf(v3 - mx);
    float inv = 1.0f / (e0 + e1 + e2 + e3);
    ms[g] = e0 * inv; ms[4 + g] = e1 * inv; ms[8 + g] = e2 * inv; ms[12 + g] = e3 * inv;
  }
}

// ---------------------------------------------------------------------------
// Integral-image corner evaluation (one thread = one channel).
// ---------------------------------------------------------------------------
__device__ __forceinline__ float sample_corner(
    const float* __restrict__ Ip,  const float* __restrict__ IXp,
    const float* __restrict__ IYp, const float* __restrict__ ITp,
    float x, float y)
{
  float Xf = ceilf(x), Yf = ceilf(y);
  int   Xi = (int)Xf,  Yi = (int)Yf;
  float dx = Xf - x,   dy = Yf - y;
  float wx1 = 0.5f * dx * dx, wx2 = dx - wx1;
  float wy1 = 0.5f * dy * dy, wy2 = dy - wy1;
  int Xm1 = Xi - 1; if (Xm1 < 0) Xm1 += H_IMG;   // numpy-style wrap for -1
  int Ym1 = Yi - 1; if (Ym1 < 0) Ym1 += W_IMG;
  const int XiW = Xi * W_IMG, Xm1W = Xm1 * W_IMG;

  float s_a  = ITp[XiW + Yi];
  float s_e1 = wy1 * (IXp[XiW + Ym1] - 0.5f * Ip[XiW + Ym1] - 0.5f * Ip[Ym1])
             + wy2 * (IXp[XiW + Yi ] - 0.5f * Ip[XiW + Yi ] - 0.5f * Ip[Yi ]);
  float s_e2 = wx1 * (IYp[Xm1W + Yi] - 0.5f * Ip[Xm1W + Yi] - 0.5f * Ip[Xm1W])
             + wx2 * (IYp[XiW  + Yi] - 0.5f * Ip[XiW  + Yi] - 0.5f * Ip[XiW ]);
  float s_c  = wx1 * wy1 * Ip[Xm1W + Ym1] + wx2 * wy1 * Ip[XiW + Ym1]
             + wx1 * wy2 * Ip[Xm1W + Yi ] + wx2 * wy2 * Ip[XiW + Yi ];
  return s_a - s_e1 - s_e2 + s_c;
}

__global__ __launch_bounds__(64) void sample_kernel(
    const float* __restrict__ I,  const float* __restrict__ IX,
    const float* __restrict__ IY, const float* __restrict__ IT,
    const float* __restrict__ coords, float* __restrict__ S_t)
{
  const int      c  = threadIdx.x;                 // channel 0..63
  const long long pt = blockIdx.x;                 // b*(M*4) + (m*4 + rect)
  const long long b  = pt / (M_PTS * 4);
  const float*   cr = coords + pt * 4;
  const float px1 = cr[0], py1 = cr[1], px2 = cr[2], py2 = cr[3];

  const size_t plane = ((size_t)b * IN_C + c) * (size_t)(H_IMG * W_IMG);
  const float* Ip  = I  + plane; const float* IXp = IX + plane;
  const float* IYp = IY + plane; const float* ITp = IT + plane;

  float s = sample_corner(Ip, IXp, IYp, ITp, px2, py2)
          - sample_corner(Ip, IXp, IYp, ITp, px1, py2)
          - sample_corner(Ip, IXp, IYp, ITp, px2, py1)
          + sample_corner(Ip, IXp, IYp, ITp, px1, py1);
  S_t[(size_t)pt * IN_C + c] = s;
}

// ---------------------------------------------------------------------------
// LayerNorm over 256 channels (one block per row), optional residual add.
// ---------------------------------------------------------------------------
__global__ __launch_bounds__(256) void ln_kernel(
    const float* __restrict__ in, const float* __restrict__ gw,
    const float* __restrict__ bw, const float* __restrict__ res,
    float* __restrict__ out)
{
  __shared__ float red[DC];
  __shared__ float s_mean, s_var;
  const int    c   = threadIdx.x;
  const size_t row = blockIdx.x;
  const float  v   = in[row * DC + c];

  red[c] = v;
  __syncthreads();
#pragma unroll
  for (int s = DC / 2; s > 0; s >>= 1) {
    if (c < s) red[c] += red[c + s];
    __syncthreads();
  }
  if (c == 0) s_mean = red[0] * (1.0f / DC);
  __syncthreads();
  const float d = v - s_mean;

  red[c] = d * d;
  __syncthreads();
#pragma unroll
  for (int s = DC / 2; s > 0; s >>= 1) {
    if (c < s) red[c] += red[c + s];
    __syncthreads();
  }
  if (c == 0) s_var = red[0] * (1.0f / DC);
  __syncthreads();

  const float rstd = rsqrtf(s_var + 1e-5f);
  float yv = d * rstd * gw[c] + bw[c];
  if (res) yv += res[row * DC + c];
  out[row * DC + c] = yv;
}

// ---------------------------------------------------------------------------
extern "C" void kernel_launch(void* const* d_in, const int* in_sizes, int n_in,
                              void* d_out, int out_size, void* d_ws, size_t ws_size,
                              hipStream_t stream)
{
  const float* x   = (const float*)d_in[0];
  const float* I   = (const float*)d_in[1];
  const float* IX  = (const float*)d_in[2];
  const float* IY  = (const float*)d_in[3];
  const float* IT  = (const float*)d_in[4];
  const float* W0  = (const float*)d_in[5];
  const float* b0  = (const float*)d_in[6];
  const float* W1  = (const float*)d_in[7];
  const float* b1  = (const float*)d_in[8];
  const float* Wp  = (const float*)d_in[9];
  const float* bp  = (const float*)d_in[10];
  const float* g1  = (const float*)d_in[11];
  const float* be1 = (const float*)d_in[12];
  const float* Wm1 = (const float*)d_in[13];
  const float* bm1 = (const float*)d_in[14];
  const float* Wm2 = (const float*)d_in[15];
  const float* bm2 = (const float*)d_in[16];
  const float* g2  = (const float*)d_in[17];
  const float* be2 = (const float*)d_in[18];
  float* out = (float*)d_out;
  (void)in_sizes; (void)n_in; (void)out_size; (void)ws_size;

  char* ws = (char*)d_ws;
  size_t off = 0;
  auto alloc = [&](size_t bytes) -> float* {
    char* p = ws + off;
    off += (bytes + 255) & ~(size_t)255;
    return (float*)p;
  };
  float* yi0    = alloc((size_t)B_SZ * 256  * 96 * 4);   // up-proj i=0
  float* yi1    = alloc((size_t)B_SZ * 768  * 96 * 4);   // up-proj i=1
  float* coords = alloc((size_t)B_SZ * M_PTS * 16 * 4);  // (b,m,rect,4)
  float* msoft  = alloc((size_t)B_SZ * M_PTS * 16 * 4);  // (b,m,s,g)
  float* S_t    = alloc((size_t)B_SZ * M_PTS * 4 * IN_C * 4);
  float* zbuf   = alloc((size_t)B_SZ * M_PTS * DC * 4);
  float* hbuf   = alloc((size_t)B_SZ * M_PTS * 4 * DC * 4);
  float* mlpbuf = S_t;  // S_t is dead after the fused Wp GEMM -> reuse

  const int npts = B_SZ * M_PTS;

  // 1) up-projection GEMMs: rows 256 / 768 per batch, K=256, N=96
  wmma_gemm_kernel<0><<<dim3(2, 4, B_SZ), 128, 0, stream>>>(
      x, (long long)N_TOK * UP_CH, UP_CH, W0, 96, b0,
      yi0, (long long)256 * 96, 96, UP_CH, 96, nullptr, nullptr);
  wmma_gemm_kernel<0><<<dim3(2, 12, B_SZ), 128, 0, stream>>>(
      x + 256 * UP_CH, (long long)N_TOK * UP_CH, UP_CH, W1, 96, b1,
      yi1, (long long)768 * 96, 96, UP_CH, 96, nullptr, nullptr);

  // 2) sigmoid coords + softmax masks
  prep_points_kernel<<<(npts + 255) / 256, 256, 0, stream>>>(yi0, yi1, coords, msoft);

  // 3) integral-image rect sampling -> S_t (B, M*4, 64)
  sample_kernel<<<B_SZ * M_PTS * 4, 64, 0, stream>>>(I, IX, IY, IT, coords, S_t);

  // 4) Wp GEMM fused with gelu + mask-weighted rect reduction -> z (B,M,256)
  wmma_gemm_kernel<2><<<dim3(4, (M_PTS * 4) / 64, B_SZ), 128, 0, stream>>>(
      S_t, (long long)M_PTS * 4 * IN_C, IN_C, Wp, DC, bp,
      nullptr, 0, 0, IN_C, DC, msoft, zbuf);

  // 5) LayerNorm 1 (in place) -> identity
  ln_kernel<<<npts, DC, 0, stream>>>(zbuf, g1, be1, nullptr, zbuf);

  // 6) MLP up: 256 -> 1024, gelu
  wmma_gemm_kernel<1><<<dim3(16, M_PTS / 64, B_SZ), 128, 0, stream>>>(
      zbuf, (long long)M_PTS * DC, DC, Wm1, 4 * DC, bm1,
      hbuf, (long long)M_PTS * 4 * DC, 4 * DC, DC, 4 * DC, nullptr, nullptr);

  // 7) MLP down: 1024 -> 256
  wmma_gemm_kernel<0><<<dim3(4, M_PTS / 64, B_SZ), 128, 0, stream>>>(
      hbuf, (long long)M_PTS * 4 * DC, 4 * DC, Wm2, DC, bm2,
      mlpbuf, (long long)M_PTS * DC, DC, 4 * DC, DC, nullptr, nullptr);

  // 8) LayerNorm 2 + residual -> output
  ln_kernel<<<npts, DC, 0, stream>>>(mlpbuf, g2, be2, zbuf, out);
}
